// DeepSignatureModel_62886911148778
// MI455X (gfx1250) — matrix operand order repaired and checked by reading, not compiled, not run on hardware
//
#include <hip/hip_runtime.h>

// ---------------------------------------------------------------------------
// DeepSignatureModel on gfx1250 (MI455X). bf16 WMMA for all augment2 matmuls,
// f32 accumulate; signature scans in LDS. See analysis above.
// ---------------------------------------------------------------------------

typedef __attribute__((ext_vector_type(16))) __bf16 v16bf;
typedef __attribute__((ext_vector_type(8)))  __bf16 v8bf;
typedef __attribute__((ext_vector_type(8)))  float  v8f;

#define NB   64      // batch
#define SS   1024    // sequence
#define CIN  5
#define H1   64
#define H2   16
#define KW   4
#define L1   1021    // S - K + 1
#define C1   22      // H2 + CIN + 1
#define SIGC1 506    // C1 + C1*C1
#define L2   1018    // L1 - K + 1
#define KBIG 2024    // SIGC1 * KW
#define KPAD 2048
#define OUTC 32
#define SIGC2 272    // H2 + H2*H2

static __device__ __forceinline__ v8f wmma_bf16(v16bf a, v16bf b, v8f c) {
  return __builtin_amdgcn_wmma_f32_16x16x32_bf16(false, a, false, b, (short)0, c,
                                                 false, false);
}

// A fragment (16x32 bf16, row-major source with rowStride elements).
// lanes 0-15: M=lane,   elems 0..15 -> K=kt*32+0..15
// lanes16-31: M=lane-16,elems 0..15 -> K=kt*32+16..31
static __device__ __forceinline__ v16bf a_frag(const __bf16* src, int m,
                                               int rowStride, int kt, int sel) {
  int base = kt * 32 + sel * 16;
  const v8bf lo = *(const v8bf*)(src + m * rowStride + base);
  const v8bf hi = *(const v8bf*)(src + m * rowStride + base + 8);
  v16bf a;
#pragma unroll
  for (int e = 0; e < 8; ++e) { a[e] = lo[e]; a[e + 8] = hi[e]; }
  return a;
}

// B fragment (32x16 bf16) from a [N][K] row-major bf16 matrix:
// lanes 0-15 hold K = kt*32+0..15 for col N=lane; lanes 16-31 hold K=kt*32+16..31.
static __device__ __forceinline__ v16bf b_frag(const __bf16* B, int n,
                                               int rowStride, int kt, int sel) {
  return *(const v16bf*)(B + n * rowStride + kt * 32 + sel * 16);
}

// ---------------------------------------------------------------------------
// k0: pack augment2 weights into bf16 WMMA B layout ([N][K], K zero padded).
// ---------------------------------------------------------------------------
__global__ void k0_pack_weights(const float* __restrict__ a2w0,
                                const float* __restrict__ a2w1,
                                const float* __restrict__ a2w2,
                                __bf16* __restrict__ B0, __bf16* __restrict__ B1,
                                __bf16* __restrict__ B2) {
  int i = blockIdx.x * blockDim.x + threadIdx.x;
  if (i < H1 * KPAD) {                    // B0: [64][2048], k = kk*506 + c
    int n = i >> 11, k = i & (KPAD - 1);
    float v = 0.0f;
    if (k < KBIG) {
      int kk = k / SIGC1;
      int c  = k - kk * SIGC1;
      v = a2w0[(n * SIGC1 + c) * KW + kk];
    }
    B0[i] = (__bf16)v;
  } else if (i < H1 * KPAD + H1 * H1) {   // B1: [64][64]
    int j = i - H1 * KPAD;
    B1[j] = (__bf16)a2w1[j];
  } else if (i < H1 * KPAD + H1 * H1 + H2 * H1) {  // B2: [16][64]
    int j = i - (H1 * KPAD + H1 * H1);
    B2[j] = (__bf16)a2w2[j];
  }
}

// ---------------------------------------------------------------------------
// k1: augment1 (3 convs + concat original/time) into LDS, then streamed
// depth-2 signature scan; emit s1 (B, 1021, 506) as bf16. One block per batch.
// ---------------------------------------------------------------------------
__global__ void k1_aug1_sig1(const float* __restrict__ xg,
                             const float* __restrict__ w0, const float* __restrict__ b0,
                             const float* __restrict__ w1, const float* __restrict__ b1,
                             const float* __restrict__ w2, const float* __restrict__ b2,
                             __bf16* __restrict__ s1g) {
  extern __shared__ char smem[];
  float* sH  = (float*)smem;            // [1021][22]
  float* sW0 = sH + L1 * C1;            // 64*5*4
  float* sB0 = sW0 + H1 * CIN * KW;     // 64
  float* sW1 = sB0 + H1;                // 64*64
  float* sB1 = sW1 + H1 * H1;           // 64
  float* sW2 = sB1 + H1;                // 16*64
  float* sB2 = sW2 + H2 * H1;           // 16

  const int tid = threadIdx.x;
  const int b   = blockIdx.x;

  for (int i = tid; i < H1 * CIN * KW; i += blockDim.x) sW0[i] = w0[i];
  for (int i = tid; i < H1;            i += blockDim.x) sB0[i] = b0[i];
  for (int i = tid; i < H1 * H1;       i += blockDim.x) sW1[i] = w1[i];
  for (int i = tid; i < H1;            i += blockDim.x) sB1[i] = b1[i];
  for (int i = tid; i < H2 * H1;       i += blockDim.x) sW2[i] = w2[i];
  for (int i = tid; i < H2;            i += blockDim.x) sB2[i] = b2[i];
  __syncthreads();

  // ---- Phase A: augment1 ----
  for (int pos = tid; pos < L1; pos += blockDim.x) {
    float xv[KW * CIN];
#pragma unroll
    for (int kk = 0; kk < KW; ++kk)
#pragma unroll
      for (int c = 0; c < CIN; ++c)
        xv[kk * CIN + c] = xg[((size_t)b * SS + pos + kk) * CIN + c];

    float v1[H1];
    for (int o = 0; o < H1; ++o) {
      float a = sB0[o];
      const float* wr = sW0 + o * (CIN * KW);
#pragma unroll
      for (int c = 0; c < CIN; ++c)
#pragma unroll
        for (int kk = 0; kk < KW; ++kk)
          a += xv[kk * CIN + c] * wr[c * KW + kk];
      v1[o] = fmaxf(a, 0.0f);           // relu before conv1
    }
    float v2[H1];
    for (int o = 0; o < H1; ++o) {
      float a = sB1[o];
      const float* wr = sW1 + o * H1;
#pragma unroll 8
      for (int i2 = 0; i2 < H1; ++i2) a += v1[i2] * wr[i2];
      v2[o] = fmaxf(a, 0.0f);           // relu before conv2
    }
    float* hr = sH + pos * C1;
#pragma unroll
    for (int c = 0; c < CIN; ++c) hr[c] = xv[3 * CIN + c];   // original (shifted K-1)
    hr[CIN] = (float)pos * (1.0f / (float)(L1 - 1));         // time channel
    for (int o = 0; o < H2; ++o) {
      float a = sB2[o];
      const float* wr = sW2 + o * H1;
#pragma unroll 8
      for (int i2 = 0; i2 < H1; ++i2) a += v2[i2] * wr[i2];
      hr[CIN + 1 + o] = a;              // no relu after last conv
    }
  }
  __syncthreads();

  // ---- Phase B: streamed signature, depth 2, basepoint=true ----
  // level1_t = h_t ; L2_t(i,j) = L2_{t-1} + h_{t-1}[i]*dx_t[j] + 0.5*dx_t[i]*dx_t[j]
  const int p0 = tid;                 // pair 0
  const int i0 = p0 / C1, j0 = p0 - i0 * C1;
  const int p1 = tid + 256;           // pair 1 (if < 484)
  const bool has1 = (p1 < C1 * C1);
  const int i1 = p1 / C1, j1 = p1 - i1 * C1;
  float a0 = 0.0f, a1 = 0.0f;

  for (int t = 0; t < L1; ++t) {
    const float* cur  = sH + t * C1;
    const float* prev = sH + (t > 0 ? (t - 1) * C1 : 0);
    const float tnz   = (t > 0) ? 1.0f : 0.0f;
    const size_t base = ((size_t)b * L1 + t) * SIGC1;

    if (tid < C1) s1g[base + tid] = (__bf16)cur[tid];

    {
      float pi = tnz * prev[i0], pj = tnz * prev[j0];
      float dxi = cur[i0] - pi, dxj = cur[j0] - pj;
      a0 += pi * dxj + 0.5f * dxi * dxj;
      s1g[base + C1 + p0] = (__bf16)a0;
    }
    if (has1) {
      float pi = tnz * prev[i1], pj = tnz * prev[j1];
      float dxi = cur[i1] - pi, dxj = cur[j1] - pj;
      a1 += pi * dxj + 0.5f * dxi * dxj;
      s1g[base + C1 + p1] = (__bf16)a1;
    }
  }
}

// ---------------------------------------------------------------------------
// k2: augment2 = bf16 WMMA GEMM chain. Grid (64 M-tiles, 64 batches),
// 128 threads = 4 waves; wave w owns output columns [16w, 16w+16).
// ---------------------------------------------------------------------------
__global__ void k2_aug2_wmma(const __bf16* __restrict__ s1g,
                             const __bf16* __restrict__ B0,
                             const __bf16* __restrict__ B1,
                             const __bf16* __restrict__ B2,
                             const float* __restrict__ bias0,
                             const float* __restrict__ bias1,
                             const float* __restrict__ bias2,
                             float* __restrict__ h2g) {
  extern __shared__ char smem[];
  __bf16* sA = (__bf16*)smem;           // [16][2048] A tile (K padded)
  __bf16* sY = sA + 16 * KPAD;          // [16][64] intermediate activations

  const int tid  = threadIdx.x;
  const int wave = tid >> 5;
  const int lane = tid & 31;
  const int t0   = blockIdx.x * 16;
  const int b    = blockIdx.y;

  // Build padded A tile: row m is [s1[t0+m], s1[t0+m+1], s1[t0+m+2], s1[t0+m+3]].
  for (int idx = tid; idx < 16 * KPAD; idx += blockDim.x) {
    int m = idx >> 11, k = idx & (KPAD - 1);
    __bf16 v = (__bf16)0.0f;
    if (k < KBIG) {
      int kk = k / SIGC1;
      int c  = k - kk * SIGC1;
      int row = t0 + m + kk;
      if (row > L1 - 1) row = L1 - 1;
      v = s1g[((size_t)b * L1 + row) * SIGC1 + c];
    }
    sA[idx] = v;
  }
  __syncthreads();

  const int sel = lane >> 4;     // K-half select
  const int nl  = lane & 15;     // local row (A) / col (B)
  const int n   = wave * 16 + nl;

  // conv0: [16 x 2048] x [2048 x 64], bias in accumulator.
  v8f acc;
  {
    float bv = bias0[n];
#pragma unroll
    for (int r = 0; r < 8; ++r) acc[r] = bv;
  }
  for (int kt = 0; kt < KPAD / 32; ++kt) {
    if (kt + 1 < KPAD / 32)
      __builtin_prefetch(B0 + n * KPAD + (kt + 1) * 32 + sel * 16, 0, 1);
    v16bf af = a_frag(sA, nl, KPAD, kt, sel);
    v16bf bf = b_frag(B0, n, KPAD, kt, sel);
    acc = wmma_bf16(af, bf, acc);
  }
#pragma unroll
  for (int r = 0; r < 8; ++r) {          // relu -> LDS bf16
    int mrow = r + sel * 8;
    sY[mrow * H1 + n] = (__bf16)fmaxf(acc[r], 0.0f);
  }
  __syncthreads();

  // conv1 (1x1): [16 x 64] x [64 x 64]
  v8f acc2;
  {
    float bv = bias1[n];
#pragma unroll
    for (int r = 0; r < 8; ++r) acc2[r] = bv;
  }
#pragma unroll
  for (int kt = 0; kt < 2; ++kt) {
    v16bf af = a_frag(sY, nl, H1, kt, sel);
    v16bf bf = b_frag(B1, n, H1, kt, sel);
    acc2 = wmma_bf16(af, bf, acc2);
  }
  __syncthreads();                        // everyone done reading sY
#pragma unroll
  for (int r = 0; r < 8; ++r) {
    int mrow = r + sel * 8;
    sY[mrow * H1 + n] = (__bf16)fmaxf(acc2[r], 0.0f);
  }
  __syncthreads();

  // conv2 (1x1): [16 x 64] x [64 x 16] -> h2, wave 0 only (no relu, has bias).
  if (wave == 0) {
    v8f acc3;
    {
      float bv = bias2[nl];
#pragma unroll
      for (int r = 0; r < 8; ++r) acc3[r] = bv;
    }
#pragma unroll
    for (int kt = 0; kt < 2; ++kt) {
      v16bf af = a_frag(sY, nl, H1, kt, sel);
      v16bf bf = b_frag(B2, nl, H1, kt, sel);
      acc3 = wmma_bf16(af, bf, acc3);
    }
#pragma unroll
    for (int r = 0; r < 8; ++r) {
      int t = t0 + r + sel * 8;
      if (t < L2) h2g[((size_t)b * L2 + t) * H2 + nl] = acc3[r];
    }
  }
}

// ---------------------------------------------------------------------------
// k3: become_constant + depth-2 signature (non-stream) + linear. One block per
// batch; 256 threads == the 16x16 level-2 pairs exactly.
// ---------------------------------------------------------------------------
__global__ void k3_sig2_linear(const float* __restrict__ h2g,
                               const int* __restrict__ lengths,
                               const float* __restrict__ lw,
                               const float* __restrict__ lb,
                               float* __restrict__ outp) {
  extern __shared__ char smem[];
  float* sH = (float*)smem;             // [1018][16]
  float* sS = sH + L2 * H2;             // [272]

  const int tid = threadIdx.x;
  const int b   = blockIdx.x;

  int adj = lengths[b] - 2 * KW + 2;
  if (adj < 1)  adj = 1;
  if (adj > L2) adj = L2;

  for (int idx = tid; idx < L2 * H2; idx += blockDim.x) {
    int t = idx >> 4, c = idx & 15;
    int tt = (t < adj) ? t : (adj - 1);
    sH[idx] = h2g[((size_t)b * L2 + tt) * H2 + c];
  }
  __syncthreads();

  const int i = tid >> 4, j = tid & 15;
  float acc = 0.0f, pi = 0.0f, pj = 0.0f;
  for (int t = 0; t < L2; ++t) {
    float ci = sH[t * H2 + i], cj = sH[t * H2 + j];
    float dxi = ci - pi, dxj = cj - pj;
    acc += pi * dxj + 0.5f * dxi * dxj;
    pi = ci; pj = cj;
  }
  sS[H2 + tid] = acc;
  if (tid < H2) sS[tid] = sH[(L2 - 1) * H2 + tid];  // level 1 = last point
  __syncthreads();

  if (tid < OUTC) {
    float a = lb[tid];
    for (int k = 0; k < SIGC2; ++k) a += sS[k] * lw[tid * SIGC2 + k];
    outp[b * OUTC + tid] = a;
  }
}

// ---------------------------------------------------------------------------
extern "C" void kernel_launch(void* const* d_in, const int* in_sizes, int n_in,
                              void* d_out, int out_size, void* d_ws, size_t ws_size,
                              hipStream_t stream) {
  const float* xg    = (const float*)d_in[0];
  const int*   lens  = (const int*)  d_in[1];
  const float* a1w0  = (const float*)d_in[2];
  const float* a1b0  = (const float*)d_in[3];
  const float* a1w1  = (const float*)d_in[4];
  const float* a1b1  = (const float*)d_in[5];
  const float* a1w2  = (const float*)d_in[6];
  const float* a1b2  = (const float*)d_in[7];
  const float* a2w0  = (const float*)d_in[8];
  const float* a2b0  = (const float*)d_in[9];
  const float* a2w1  = (const float*)d_in[10];
  const float* a2b1  = (const float*)d_in[11];
  const float* a2w2  = (const float*)d_in[12];
  const float* a2b2  = (const float*)d_in[13];
  const float* linw  = (const float*)d_in[14];
  const float* linb  = (const float*)d_in[15];
  float* outp = (float*)d_out;

  // Workspace layout (bytes).
  char* ws = (char*)d_ws;
  const size_t off_s1 = 0;                                          // bf16 64*1021*506
  const size_t off_B0 = off_s1 + (size_t)NB * L1 * SIGC1 * 2;       // 66,128,128
  const size_t off_B1 = off_B0 + (size_t)H1 * KPAD * 2;             // +262,144
  const size_t off_B2 = off_B1 + (size_t)H1 * H1 * 2;               // +8,192
  const size_t off_h2 = off_B2 + (size_t)H2 * H1 * 2;               // +2,048
  __bf16* s1g = (__bf16*)(ws + off_s1);
  __bf16* B0  = (__bf16*)(ws + off_B0);
  __bf16* B1  = (__bf16*)(ws + off_B1);
  __bf16* B2  = (__bf16*)(ws + off_B2);
  float*  h2g = (float*) (ws + off_h2);

  const int lds_k1 = (L1 * C1 + H1 * CIN * KW + H1 + H1 * H1 + H1 + H2 * H1 + H2) * 4;
  const int lds_k2 = (16 * KPAD + 16 * H1) * 2;
  const int lds_k3 = (L2 * H2 + SIGC2) * 4;
  (void)hipFuncSetAttribute((const void*)k1_aug1_sig1,
                            hipFuncAttributeMaxDynamicSharedMemorySize, lds_k1);
  (void)hipFuncSetAttribute((const void*)k2_aug2_wmma,
                            hipFuncAttributeMaxDynamicSharedMemorySize, lds_k2);
  (void)hipFuncSetAttribute((const void*)k3_sig2_linear,
                            hipFuncAttributeMaxDynamicSharedMemorySize, lds_k3);

  // k0: pack weights to bf16 WMMA layout.
  {
    int total = H1 * KPAD + H1 * H1 + H2 * H1;
    k0_pack_weights<<<(total + 255) / 256, 256, 0, stream>>>(a2w0, a2w1, a2w2,
                                                             B0, B1, B2);
  }
  // k1: augment1 + streamed signature per batch.
  k1_aug1_sig1<<<NB, 256, lds_k1, stream>>>(xg, a1w0, a1b0, a1w1, a1b1,
                                            a1w2, a1b2, s1g);
  // k2: augment2 as WMMA GEMM chain.
  {
    dim3 grid((L2 + 15) / 16, NB);
    k2_aug2_wmma<<<grid, 128, lds_k2, stream>>>(s1g, B0, B1, B2,
                                                a2b0, a2b1, a2b2, h2g);
  }
  // k3: become_constant + final signature + linear.
  k3_sig2_linear<<<NB, 256, lds_k3, stream>>>(h2g, lens, linw, linb, outp);
}
